// CrossAttentionLayer_51376398795339
// MI455X (gfx1250) — matrix operand order repaired and checked by reading, not compiled, not run on hardware
//
#include <hip/hip_runtime.h>
#include <hip/hip_bf16.h>

typedef unsigned int  uint32;
typedef unsigned short ushort16;
typedef int v4i __attribute__((vector_size(16)));

typedef __attribute__((ext_vector_type(16))) __bf16 v16bf;
typedef __attribute__((ext_vector_type(8)))  float  v8f;

union BF16Frag { v16bf v; uint32 u[8]; };

#define B_  2
#define LQ_ 1024
#define LKV_ 4096
#define H_  1024
#define NH_ 16
#define HD_ 64

#define AS1 __attribute__((address_space(1)))
#define AS3 __attribute__((address_space(3)))

#if defined(__has_builtin)
# if __has_builtin(__builtin_amdgcn_global_load_async_to_lds_b128)
#  define HAVE_ASYNC_LDS 1
# endif
# if __has_builtin(__builtin_amdgcn_s_wait_asynccnt)
#  define HAVE_WAIT_ASYNC 1
# endif
#endif

__device__ __forceinline__ void wait_async_zero() {
#if defined(HAVE_ASYNC_LDS)
# if defined(HAVE_WAIT_ASYNC)
    __builtin_amdgcn_s_wait_asynccnt(0);
# else
    asm volatile("s_wait_asynccnt 0x0" ::: "memory");
# endif
#endif
}

__device__ __forceinline__ uint32 f2bf_u(float f) {
    union { float f; uint32 u; } x; x.f = f;
    return (x.u + 0x7FFFu + ((x.u >> 16) & 1u)) >> 16;   // RNE truncate to bf16
}
__device__ __forceinline__ ushort16 f2bf(float f) { return (ushort16)f2bf_u(f); }
__device__ __forceinline__ uint32 pack2(float lo, float hi) {
    return f2bf_u(lo) | (f2bf_u(hi) << 16);
}

// ---------------------------------------------------------------------------
// GEMM: Y[M,N] = X[M,K] @ W[N,K]^T + bias, optional +residual.
// OUT_F32=false -> bf16 output (Q/K/V projections into workspace)
// OUT_F32=true  -> f32 output + residual (O projection)
// Block: 128 threads (4 waves). Tile 64x64, K-step 32. bf16 WMMA, f32 acc.
// ---------------------------------------------------------------------------
template <bool OUT_F32>
__global__ __launch_bounds__(128)
void gemm_xwT_bias(const float* __restrict__ X, const float* __restrict__ W,
                   const float* __restrict__ bias, int M, int N, int K,
                   ushort16* __restrict__ outB, float* __restrict__ outF,
                   const float* __restrict__ res) {
    constexpr int STR = 40;                 // 32 + 8 pad (ushorts)
    __shared__ ushort16 ldsX[64 * STR];
    __shared__ ushort16 ldsW[64 * STR];

    const int tid  = threadIdx.x;
    const int lane = tid & 31;
    const int wv   = tid >> 5;
    const int r    = lane & 15;
    const int hi   = lane >> 4;
    const int m0   = blockIdx.x * 64;
    const int n0   = blockIdx.y * 64;

    const int lrow = tid >> 1;              // 0..63
    const int lh   = tid & 1;               // half-row of 16 floats

    v8f acc[4] = {};

    for (int k0 = 0; k0 < K; k0 += 32) {
        __syncthreads();
        // stage X tile (64x32) as bf16
        {
            const float4* src = (const float4*)(X + (size_t)(m0 + lrow) * K + k0 + lh * 16);
            uint32* dst = (uint32*)&ldsX[lrow * STR + lh * 16];
            #pragma unroll
            for (int j = 0; j < 4; ++j) {
                float4 v = src[j];
                dst[2 * j]     = pack2(v.x, v.y);
                dst[2 * j + 1] = pack2(v.z, v.w);
            }
        }
        // stage W tile (64x32) as bf16
        {
            const float4* src = (const float4*)(W + (size_t)(n0 + lrow) * K + k0 + lh * 16);
            uint32* dst = (uint32*)&ldsW[lrow * STR + lh * 16];
            #pragma unroll
            for (int j = 0; j < 4; ++j) {
                float4 v = src[j];
                dst[2 * j]     = pack2(v.x, v.y);
                dst[2 * j + 1] = pack2(v.z, v.w);
            }
        }
        // prefetch next K-chunk while WMMAs run (global_prefetch_b8)
        if (k0 + 32 < K) {
            __builtin_prefetch((const void*)(X + (size_t)(m0 + lrow) * K + k0 + 32 + lh * 16), 0, 1);
            __builtin_prefetch((const void*)(W + (size_t)(n0 + lrow) * K + k0 + 32 + lh * 16), 0, 1);
        }
        __syncthreads();

        // A fragment: rows wv*16 + r, K-chunk layout per ISA 16-bit A 16x32
        BF16Frag a;
        {
            const uint32* p1 = (const uint32*)&ldsX[(wv * 16 + r) * STR + hi * 8];
            const uint32* p2 = (const uint32*)&ldsX[(wv * 16 + r) * STR + hi * 8 + 16];
            #pragma unroll
            for (int j = 0; j < 4; ++j) { a.u[j] = p1[j]; a.u[4 + j] = p2[j]; }
        }
        #pragma unroll
        for (int n = 0; n < 4; ++n) {
            BF16Frag bf;
            const uint32* pb = (const uint32*)&ldsW[(n * 16 + r) * STR + hi * 8];
            #pragma unroll
            for (int j = 0; j < 4; ++j) { bf.u[j] = pb[j]; bf.u[4 + j] = pb[8 + j]; }
            acc[n] = __builtin_amdgcn_wmma_f32_16x16x32_bf16(
                false, a.v, false, bf.v, (short)0, acc[n], false, false);
        }
    }

    #pragma unroll
    for (int n = 0; n < 4; ++n) {
        const int col = n0 + n * 16 + r;
        const float bc = bias[col];
        #pragma unroll
        for (int i = 0; i < 8; ++i) {
            const int row = m0 + wv * 16 + i + 8 * hi;
            float v = acc[n][i] + bc;
            if (OUT_F32) {
                outF[(size_t)row * N + col] = v + res[(size_t)row * N + col];
            } else {
                outB[(size_t)row * N + col] = f2bf(v);
            }
        }
    }
}

// ---------------------------------------------------------------------------
// Attention: per (b, h, 64 q-rows). 4 waves, each owns a 16-row q strip.
// K tile staged via ASYNC global->LDS copies (when toolchain exposes them);
// V tile transposed through registers; online softmax; writes masked
// pre-softmax scores (output #2) and ctx (ws).
// ---------------------------------------------------------------------------
__global__ __launch_bounds__(128)
void attn_kernel(const ushort16* __restrict__ qb, const ushort16* __restrict__ kb,
                 const ushort16* __restrict__ vb, const int* __restrict__ mask,
                 float* __restrict__ scores, float* __restrict__ ctx) {
    constexpr int STK = 88;                 // 64 + 24 pad; 176B row stride (16B-aligned)
    __shared__ ushort16 ldsK[64 * STK];     // K tile  [kv][hd]
    __shared__ ushort16 ldsVT[64 * STK];    // V tile transposed [hd][kv]
    __shared__ ushort16 ldsP[64 * STK];     // P re-pack, 16 rows per wave

    const int tid  = threadIdx.x;
    const int lane = tid & 31;
    const int wv   = tid >> 5;
    const int r    = lane & 15;
    const int hi   = lane >> 4;
    const int h    = blockIdx.y;
    const int b    = blockIdx.z;
    const int q0   = blockIdx.x * 64 + wv * 16;

    // Q fragments for both 32-wide K-dim chunks (HD=64)
    BF16Frag aQ[2];
    {
        const ushort16* qrow = qb + (size_t)(b * LQ_ + q0 + r) * H_ + h * HD_;
        #pragma unroll
        for (int c = 0; c < 2; ++c) {
            const uint32* p1 = (const uint32*)(qrow + c * 32 + hi * 8);
            const uint32* p2 = (const uint32*)(qrow + c * 32 + hi * 8 + 16);
            #pragma unroll
            for (int j = 0; j < 4; ++j) { aQ[c].u[j] = p1[j]; aQ[c].u[4 + j] = p2[j]; }
        }
    }

    float m_i[8], l_i[8];
    v8f o[4] = {};
    #pragma unroll
    for (int i = 0; i < 8; ++i) { m_i[i] = -INFINITY; l_i[i] = 0.f; }

    const int lrow = tid >> 1;              // 0..63 kv row within tile
    const int lh   = tid & 1;               // half of HD

    for (int kv0 = 0; kv0 < LKV_; kv0 += 64) {
        __syncthreads();
        // ---- stage K tile: async DMA copy global(bf16) -> LDS when available
        {
            const ushort16* ksrc =
                kb + (size_t)(b * LKV_ + kv0 + lrow) * H_ + h * HD_ + lh * 32;
#if defined(HAVE_ASYNC_LDS)
            #pragma unroll
            for (int j = 0; j < 4; ++j) {
                __builtin_amdgcn_global_load_async_to_lds_b128(
                    (AS1 v4i*)(ksrc + j * 8),
                    (AS3 v4i*)&ldsK[lrow * STK + lh * 32 + j * 8], 0, 0);
            }
#else
            const uint4* ks = (const uint4*)ksrc;
            uint32* kd = (uint32*)&ldsK[lrow * STK + lh * 32];
            #pragma unroll
            for (int j = 0; j < 4; ++j) {
                uint4 v = ks[j];
                kd[4 * j] = v.x; kd[4 * j + 1] = v.y; kd[4 * j + 2] = v.z; kd[4 * j + 3] = v.w;
            }
#endif
        }
        // ---- stage V tile transposed (needs register pass)
        {
            const uint4* vs = (const uint4*)(vb + (size_t)(b * LKV_ + kv0 + lrow) * H_ + h * HD_ + lh * 32);
            #pragma unroll
            for (int j = 0; j < 4; ++j) {
                uint4 v = vs[j];
                const int cb = lh * 32 + j * 8;
                ldsVT[(cb + 0) * STK + lrow] = (ushort16)(v.x & 0xFFFF);
                ldsVT[(cb + 1) * STK + lrow] = (ushort16)(v.x >> 16);
                ldsVT[(cb + 2) * STK + lrow] = (ushort16)(v.y & 0xFFFF);
                ldsVT[(cb + 3) * STK + lrow] = (ushort16)(v.y >> 16);
                ldsVT[(cb + 4) * STK + lrow] = (ushort16)(v.z & 0xFFFF);
                ldsVT[(cb + 5) * STK + lrow] = (ushort16)(v.z >> 16);
                ldsVT[(cb + 6) * STK + lrow] = (ushort16)(v.w & 0xFFFF);
                ldsVT[(cb + 7) * STK + lrow] = (ushort16)(v.w >> 16);
            }
        }
        // ---- prefetch next tile's K/V lines into cache while we compute
        if (kv0 + 64 < LKV_) {
            __builtin_prefetch(
                (const void*)(kb + (size_t)(b * LKV_ + kv0 + 64 + lrow) * H_ + h * HD_ + lh * 32), 0, 1);
            __builtin_prefetch(
                (const void*)(vb + (size_t)(b * LKV_ + kv0 + 64 + lrow) * H_ + h * HD_ + lh * 32), 0, 1);
        }
        wait_async_zero();
        __syncthreads();

        // ---- S = Q K^T over four 16-col subtiles
        v8f sT[4];
        #pragma unroll
        for (int n = 0; n < 4; ++n) {
            v8f s = {};
            #pragma unroll
            for (int c = 0; c < 2; ++c) {
                BF16Frag bk;
                const uint32* pb = (const uint32*)&ldsK[(n * 16 + r) * STK + c * 32 + hi * 8];
                #pragma unroll
                for (int j = 0; j < 4; ++j) { bk.u[j] = pb[j]; bk.u[4 + j] = pb[8 + j]; }
                s = __builtin_amdgcn_wmma_f32_16x16x32_bf16(
                    false, aQ[c].v, false, bk.v, (short)0, s, false, false);
            }
            const int kvcol = kv0 + n * 16 + r;
            const int mv = mask[b * LKV_ + kvcol];
            #pragma unroll
            for (int i = 0; i < 8; ++i) {
                float vsc = s[i] * 0.125f;                // 1/sqrt(64)
                if (mv == 0) vsc = -INFINITY;
                s[i] = vsc;
                const int rq = q0 + i + 8 * hi;
                scores[((size_t)(b * NH_ + h) * LQ_ + rq) * LKV_ + kvcol] = vsc;
            }
            sT[n] = s;
        }

        // ---- online softmax per row (16-lane groups == D fragment rows)
        #pragma unroll
        for (int i = 0; i < 8; ++i) {
            float tm = fmaxf(fmaxf(sT[0][i], sT[1][i]), fmaxf(sT[2][i], sT[3][i]));
            #pragma unroll
            for (int mskw = 1; mskw < 16; mskw <<= 1) tm = fmaxf(tm, __shfl_xor(tm, mskw));
            const float mn = fmaxf(m_i[i], tm);
            float f = 1.f, ps = 0.f;
            if (mn == -INFINITY) {
                #pragma unroll
                for (int n = 0; n < 4; ++n) sT[n][i] = 0.f;
            } else {
                f = __expf(m_i[i] - mn);
                #pragma unroll
                for (int n = 0; n < 4; ++n) {
                    float pv = __expf(sT[n][i] - mn);
                    sT[n][i] = pv; ps += pv;
                }
            }
            #pragma unroll
            for (int mskw = 1; mskw < 16; mskw <<= 1) ps += __shfl_xor(ps, mskw);
            l_i[i] = l_i[i] * f + ps;
            m_i[i] = mn;
            #pragma unroll
            for (int n = 0; n < 4; ++n) o[n][i] *= f;
        }

        // ---- re-pack P through per-wave LDS into A-fragment layout
        #pragma unroll
        for (int n = 0; n < 4; ++n)
            #pragma unroll
            for (int i = 0; i < 8; ++i)
                ldsP[(wv * 16 + i + 8 * hi) * STK + n * 16 + r] = f2bf(sT[n][i]);

        BF16Frag aP[2];
        #pragma unroll
        for (int c = 0; c < 2; ++c) {
            const uint32* p1 = (const uint32*)&ldsP[(wv * 16 + r) * STK + c * 32 + hi * 8];
            const uint32* p2 = (const uint32*)&ldsP[(wv * 16 + r) * STK + c * 32 + hi * 8 + 16];
            #pragma unroll
            for (int j = 0; j < 4; ++j) { aP[c].u[j] = p1[j]; aP[c].u[4 + j] = p2[j]; }
        }
        // ---- ctx += P @ V   (B fragment reads contiguous rows of V^T)
        #pragma unroll
        for (int n = 0; n < 4; ++n) {
            #pragma unroll
            for (int c = 0; c < 2; ++c) {
                BF16Frag bv;
                const uint32* pb = (const uint32*)&ldsVT[(n * 16 + r) * STK + c * 32 + hi * 8];
                #pragma unroll
                for (int j = 0; j < 4; ++j) { bv.u[j] = pb[j]; bv.u[4 + j] = pb[8 + j]; }
                o[n] = __builtin_amdgcn_wmma_f32_16x16x32_bf16(
                    false, aP[c].v, false, bv.v, (short)0, o[n], false, false);
            }
        }
    }

    #pragma unroll
    for (int i = 0; i < 8; ++i) {
        const float rl = (l_i[i] > 0.f) ? (1.f / l_i[i]) : 0.f;
        const int rq = q0 + i + 8 * hi;
        #pragma unroll
        for (int n = 0; n < 4; ++n)
            ctx[(size_t)(b * LQ_ + rq) * H_ + h * HD_ + n * 16 + r] = o[n][i] * rl;
    }
}

// ---------------------------------------------------------------------------
// Row LayerNorm over H=1024, one row per block (256 threads)
// ---------------------------------------------------------------------------
__global__ __launch_bounds__(256)
void ln_kernel(const float* __restrict__ y, const float* __restrict__ g,
               const float* __restrict__ bta, float* __restrict__ out) {
    __shared__ float red[256];
    const int row = blockIdx.x;
    const float* yr = y + (size_t)row * H_;
    float s = 0.f;
    for (int j = threadIdx.x; j < H_; j += 256) s += yr[j];
    red[threadIdx.x] = s; __syncthreads();
    for (int st = 128; st > 0; st >>= 1) {
        if (threadIdx.x < st) red[threadIdx.x] += red[threadIdx.x + st];
        __syncthreads();
    }
    const float mu = red[0] * (1.f / H_);
    __syncthreads();
    float vs = 0.f;
    for (int j = threadIdx.x; j < H_; j += 256) { float d = yr[j] - mu; vs += d * d; }
    red[threadIdx.x] = vs; __syncthreads();
    for (int st = 128; st > 0; st >>= 1) {
        if (threadIdx.x < st) red[threadIdx.x] += red[threadIdx.x + st];
        __syncthreads();
    }
    const float inv = rsqrtf(red[0] * (1.f / H_) + 1e-12f);
    for (int j = threadIdx.x; j < H_; j += 256)
        out[(size_t)row * H_ + j] = (yr[j] - mu) * inv * g[j] + bta[j];
}

// ---------------------------------------------------------------------------
extern "C" void kernel_launch(void* const* d_in, const int* in_sizes, int n_in,
                              void* d_out, int out_size, void* d_ws, size_t ws_size,
                              hipStream_t stream) {
    const float* hidden = (const float*)d_in[0];
    const float* enc    = (const float*)d_in[1];
    const int*   mask   = (const int*)d_in[2];
    const float* q_w = (const float*)d_in[3];
    const float* q_b = (const float*)d_in[4];
    const float* k_w = (const float*)d_in[5];
    const float* k_b = (const float*)d_in[6];
    const float* v_w = (const float*)d_in[7];
    const float* v_b = (const float*)d_in[8];
    const float* o_w = (const float*)d_in[9];
    const float* o_b = (const float*)d_in[10];
    const float* ln_w = (const float*)d_in[11];
    const float* ln_b = (const float*)d_in[12];

    char* ws = (char*)d_ws;
    ushort16* qbf = (ushort16*)(ws);                         //  4 MiB
    ushort16* kbf = (ushort16*)(ws + ((size_t)4  << 20));    // 16 MiB
    ushort16* vbf = (ushort16*)(ws + ((size_t)20 << 20));    // 16 MiB
    float*    yf  = (float*)(ws + ((size_t)36 << 20));       //  8 MiB
    float*    ctx = (float*)(ws + ((size_t)44 << 20));       //  8 MiB

    float* out    = (float*)d_out;
    float* scores = out + (size_t)B_ * LQ_ * H_;

    const dim3 blk(128);
    // Q / K / V projections (bf16 outputs)
    gemm_xwT_bias<false><<<dim3((B_ * LQ_) / 64, H_ / 64), blk, 0, stream>>>(
        hidden, q_w, q_b, B_ * LQ_, H_, H_, qbf, nullptr, nullptr);
    gemm_xwT_bias<false><<<dim3((B_ * LKV_) / 64, H_ / 64), blk, 0, stream>>>(
        enc, k_w, k_b, B_ * LKV_, H_, H_, kbf, nullptr, nullptr);
    gemm_xwT_bias<false><<<dim3((B_ * LKV_) / 64, H_ / 64), blk, 0, stream>>>(
        enc, v_w, v_b, B_ * LKV_, H_, H_, vbf, nullptr, nullptr);
    // attention (writes masked scores + ctx)
    attn_kernel<<<dim3(LQ_ / 64, NH_, B_), blk, 0, stream>>>(
        qbf, kbf, vbf, mask, scores, ctx);
    // output projection + bias + residual -> y
    gemm_xwT_bias<true><<<dim3((B_ * LQ_) / 64, H_ / 64), blk, 0, stream>>>(
        ctx, o_w, o_b, B_ * LQ_, H_, H_, nullptr, yf, hidden);
    // layernorm -> out
    ln_kernel<<<B_ * LQ_, 256, 0, stream>>>(yf, ln_w, ln_b, out);
}
